// MultiHeadAttention_87608742904148
// MI455X (gfx1250) — matrix operand order repaired and checked by reading, not compiled, not run on hardware
//
#include <hip/hip_runtime.h>
#include <math.h>

// Problem constants (from reference): B=4, T=2048, D=1024, H=16, HD=64
constexpr int B_  = 4;
constexpr int T_  = 2048;
constexpr int D_  = 1024;
constexpr int H_  = 16;
constexpr int HD_ = 64;

typedef __attribute__((ext_vector_type(16))) __bf16 v16bf;
typedef __attribute__((ext_vector_type(8)))  __bf16 v8bf;
typedef __attribute__((ext_vector_type(4)))  __bf16 v4bf;
typedef __attribute__((ext_vector_type(8)))  float  v8f;
typedef __attribute__((ext_vector_type(4)))  float  v4f;

__device__ __forceinline__ v8f wmma_bf16(v16bf a, v16bf b, v8f c) {
  // 8 args: (neg_a, A, neg_b, B, c_mod, C, reuse_a, reuse_b)
  return __builtin_amdgcn_wmma_f32_16x16x32_bf16(false, a, false, b, (short)0, c,
                                                 false, false);
}

// A-operand (16x32 bf16, M x K). base points at (row0, k0), row-major, stride ld.
// Lanes 0-15 / 16-31 both hold M=lane&15; lane-half selects K-halves:
// e<8 -> K = half*8 + e ; e>=8 -> K = 16 + half*8 + (e-8)
__device__ __forceinline__ v16bf load_a(const __bf16* base, int ld, int lane) {
  int m = lane & 15, half = lane >> 4;
  const __bf16* p = base + (size_t)m * ld + half * 8;
  v8bf lo = *(const v8bf*)(p);
  v8bf hi = *(const v8bf*)(p + 16);
  v16bf a;
#pragma unroll
  for (int e = 0; e < 8; ++e) { a[e] = lo[e]; a[e + 8] = hi[e]; }
  return a;
}

// B-operand (32x16 bf16, K x N) where B[k][n] = base[n*ld + k]
// (source row-major along K, one row per output column). Two 16B loads per lane,
// same K order as load_a: e<8 -> K = half*8 + e ; e>=8 -> K = 16 + half*8 + (e-8)
__device__ __forceinline__ v16bf load_b(const __bf16* base, int ld, int lane) {
  const __bf16* p = base + (size_t)(lane & 15) * ld + (lane >> 4) * 8;
  v8bf lo = *(const v8bf*)(p);
  v8bf hi = *(const v8bf*)(p + 16);
  v16bf b;
#pragma unroll
  for (int e = 0; e < 8; ++e) { b[e] = lo[e]; b[e + 8] = hi[e]; }
  return b;
}

// A-operand built from the fp32 LDS score slab (row stride T_), two v8f LDS loads.
__device__ __forceinline__ v16bf slab_a(const float* slab, int k0, int lane) {
  int m = lane & 15, half = lane >> 4;
  const float* p = slab + m * T_ + k0 + half * 8;
  v8f lo = *(const v8f*)(p);
  v8f hi = *(const v8f*)(p + 16);
  v16bf a;
#pragma unroll
  for (int e = 0; e < 8; ++e) { a[e] = (__bf16)lo[e]; a[e + 8] = (__bf16)hi[e]; }
  return a;
}

// 16x64 GEMM tile over K=D_: uniform two-stage software pipeline (2x unrolled,
// no epilogue). The final prefetch wraps to k=0 ((k0+64)&(D_-1)) so every load
// writes directly into the loop-carried registers -> no rotation moves.
__device__ __forceinline__ void gemm_16x64(const __bf16* __restrict__ xrow,
                                           const __bf16* __restrict__ wbase,
                                           int n0, int lane, v8f acc[4]) {
  v16bf a0, a1, b0[4], b1[4];
  a0 = load_a(xrow, D_, lane);
#pragma unroll
  for (int j = 0; j < 4; ++j)
    b0[j] = load_b(wbase + (size_t)(n0 + j * 16) * D_, D_, lane);

  for (int k0 = 0; k0 < D_; k0 += 64) {
    int k1 = k0 + 32;               // always < D_
    int k2 = (k0 + 64) & (D_ - 1);  // wraps to 0 on last iteration (dummy load)
    a1 = load_a(xrow + k1, D_, lane);
#pragma unroll
    for (int j = 0; j < 4; ++j)
      b1[j] = load_b(wbase + (size_t)(n0 + j * 16) * D_ + k1, D_, lane);
#pragma unroll
    for (int j = 0; j < 4; ++j) acc[j] = wmma_bf16(a0, b0[j], acc[j]);

    a0 = load_a(xrow + k2, D_, lane);
#pragma unroll
    for (int j = 0; j < 4; ++j)
      b0[j] = load_b(wbase + (size_t)(n0 + j * 16) * D_ + k2, D_, lane);
#pragma unroll
    for (int j = 0; j < 4; ++j) acc[j] = wmma_bf16(a1, b1[j], acc[j]);
  }
}

// ---------------------------------------------------------------- convert f32->bf16
__global__ void cvt_f32_to_bf16(const float* __restrict__ src,
                                __bf16* __restrict__ dst, int n4) {
  int i = blockIdx.x * blockDim.x + threadIdx.x;
  if (i < n4) {
    v4f v = ((const v4f*)src)[i];
    v4bf o;
    o[0] = (__bf16)v[0]; o[1] = (__bf16)v[1];
    o[2] = (__bf16)v[2]; o[3] = (__bf16)v[3];
    ((v4bf*)dst)[i] = o;
  }
}

// ---------------------------------------------------------------- QKV projection
// Y = X @ W^T ; X:[B*T, D] bf16, W:[D, D] bf16 (torch [out,in] layout => B-operand
// reads W row n along k contiguously). Each wave computes a 16x64 tile.
// Q,K stored [B,H,T,HD]; V stored transposed [B,H,HD,T] for the P@V B-operand.
__global__ void __launch_bounds__(256)
proj_qkv_kernel(const __bf16* __restrict__ x,  const __bf16* __restrict__ wq,
                const __bf16* __restrict__ wk, const __bf16* __restrict__ wv,
                __bf16* __restrict__ qo, __bf16* __restrict__ ko,
                __bf16* __restrict__ vt) {
  int lane = threadIdx.x & 31, wave = threadIdx.x >> 5;
  int m0 = blockIdx.x * 16;                 // row tile in [0, B*T)
  int n0 = blockIdx.y * 512 + wave * 64;    // col tile in [0, D)
  int which = blockIdx.z;                   // 0=Q 1=K 2=V
  const __bf16* w = (which == 0) ? wq : (which == 1) ? wk : wv;

  v8f acc[4] = {};
  gemm_16x64(x + (size_t)m0 * D_, w, n0, lane, acc);

  int colb = lane & 15, rowgrp = lane >> 4;
  int bidx = m0 >> 11;            // m0 / T
  int t0   = m0 & (T_ - 1);
#pragma unroll
  for (int j = 0; j < 4; ++j) {
    int n = n0 + j * 16 + colb;
    int h = n >> 6, d = n & 63;
    size_t bh = (size_t)(bidx * H_ + h);
#pragma unroll
    for (int r = 0; r < 8; ++r) {
      int t = t0 + r + 8 * rowgrp;
      __bf16 val = (__bf16)acc[j][r];
      if (which == 0)      qo[(bh * T_ + t) * HD_ + d] = val;
      else if (which == 1) ko[(bh * T_ + t) * HD_ + d] = val;
      else                 vt[(bh * HD_ + d) * T_ + t] = val;
    }
  }
}

// ---------------------------------------------------------------- attention core
// One workgroup (4 waves) per (b, h, 16-row q tile). Score slab 16x2048 fp32 in LDS.
__global__ void __launch_bounds__(128)
attn_kernel(const __bf16* __restrict__ qb, const __bf16* __restrict__ kb,
            const __bf16* __restrict__ vt, float* __restrict__ attn,
            __bf16* __restrict__ ctx) {
  __shared__ __align__(128) float slab[16 * T_];  // 128 KB (CDNA5: 320 KB/WG max)
  __shared__ float red[128];

  int tid  = threadIdx.x;
  int lane = tid & 31, wave = tid >> 5;
  int qt = blockIdx.x, h = blockIdx.y, b = blockIdx.z;
  int q0 = qt * 16;
  size_t bh = (size_t)(b * H_ + h);

  // Q tile A-operands (K = 0..31 and 32..63), reused across all k-tiles
  const __bf16* qbase = qb + (bh * T_ + q0) * HD_;
  v16bf aq0 = load_a(qbase,      HD_, lane);
  v16bf aq1 = load_a(qbase + 32, HD_, lane);

  // --- scores: waves stride over causal k-tiles, pipelined B prefetch
  int kt = wave;
  if (kt <= qt) {
    const __bf16* kbase = kb + (bh * T_ + kt * 16) * HD_;
    v16bf b0 = load_b(kbase,      HD_, lane);
    v16bf b1 = load_b(kbase + 32, HD_, lane);
    for (int ktn = kt + 4; ktn <= qt; ktn += 4) {
      const __bf16* kb2 = kb + (bh * T_ + ktn * 16) * HD_;
      v16bf nb0 = load_b(kb2,      HD_, lane);
      v16bf nb1 = load_b(kb2 + 32, HD_, lane);
      v8f acc = {};
      acc = wmma_bf16(aq0, b0, acc);
      acc = wmma_bf16(aq1, b1, acc);
      int col = kt * 16 + (lane & 15);
#pragma unroll
      for (int r = 0; r < 8; ++r)
        slab[(r + 8 * (lane >> 4)) * T_ + col] = acc[r] * 0.125f;  // 1/sqrt(HD)
      kt = ktn; b0 = nb0; b1 = nb1;
    }
    v8f acc = {};
    acc = wmma_bf16(aq0, b0, acc);
    acc = wmma_bf16(aq1, b1, acc);
    int col = kt * 16 + (lane & 15);
#pragma unroll
    for (int r = 0; r < 8; ++r)
      slab[(r + 8 * (lane >> 4)) * T_ + col] = acc[r] * 0.125f;
  }
  __syncthreads();

  int kpad = 32 * ((16 * (qt + 1) + 31) / 32);  // keys rounded up for PV chunks

  // --- softmax: 8 threads per row
  int row = tid >> 3, sub = tid & 7;
  int qidx = q0 + row;
  float pmax = -INFINITY;
  for (int k = sub; k <= qidx; k += 8) pmax = fmaxf(pmax, slab[row * T_ + k]);
  red[tid] = pmax;                                        __syncthreads();
  if (sub < 4)  red[tid] = fmaxf(red[tid], red[tid + 4]); __syncthreads();
  if (sub < 2)  red[tid] = fmaxf(red[tid], red[tid + 2]); __syncthreads();
  if (sub == 0) red[tid] = fmaxf(red[tid], red[tid + 1]); __syncthreads();
  float rmax = red[row * 8];
  __syncthreads();

  float psum = 0.f;
  for (int k = sub; k <= qidx; k += 8) psum += __expf(slab[row * T_ + k] - rmax);
  red[tid] = psum;                                        __syncthreads();
  if (sub < 4)  red[tid] += red[tid + 4];                 __syncthreads();
  if (sub < 2)  red[tid] += red[tid + 2];                 __syncthreads();
  if (sub == 0) red[tid] += red[tid + 1];                 __syncthreads();
  float rinv = 1.0f / red[row * 8];

  for (int k = sub; k <= qidx; k += 8)
    slab[row * T_ + k] = __expf(slab[row * T_ + k] - rmax) * rinv;
  for (int k = qidx + 1 + sub; k < kpad; k += 8)  // zero masked/pad cols for PV
    slab[row * T_ + k] = 0.f;
  __syncthreads();

  // --- write attn [b,h,q,k] (zeros above diagonal): v4f non-temporal streaming
  // (1.07 GB total output; NT keeps it out of the 192 MB L2 so K/V stay resident)
  float* aout = attn + (bh * T_ + q0) * (size_t)T_;
  for (int idx = tid; idx < 16 * (T_ / 4); idx += 128) {
    int r = idx >> 9;                 // T_/4 = 512 quads per row
    int k = (idx & 511) * 4;
    int q = q0 + r;
    const float* s = &slab[r * T_ + k];
    v4f v;
    v[0] = (k     <= q) ? s[0] : 0.f;
    v[1] = (k + 1 <= q) ? s[1] : 0.f;
    v[2] = (k + 2 <= q) ? s[2] : 0.f;
    v[3] = (k + 3 <= q) ? s[3] : 0.f;
    __builtin_nontemporal_store(v, (v4f*)&aout[(size_t)r * T_ + k]);
  }

  // --- ctx = P @ V : each wave owns one 16-wide d-tile, pipelined B prefetch
  int d0 = wave * 16;
  int nchunks = kpad / 32;
  const __bf16* vbase = vt + (bh * HD_ + d0) * T_;
  v8f acc = {};
  v16bf bb = load_b(vbase, T_, lane);
  for (int c = 1; c < nchunks; ++c) {
    v16bf bn = load_b(vbase + c * 32, T_, lane);
    v16bf a = slab_a(slab, (c - 1) * 32, lane);
    acc = wmma_bf16(a, bb, acc);
    bb = bn;
  }
  {
    v16bf a = slab_a(slab, (nchunks - 1) * 32, lane);
    acc = wmma_bf16(a, bb, acc);
  }
  int colb = lane & 15, rowgrp = lane >> 4;
#pragma unroll
  for (int r = 0; r < 8; ++r) {
    int t = q0 + r + 8 * rowgrp;
    ctx[((size_t)b * T_ + t) * D_ + h * HD_ + d0 + colb] = (__bf16)acc[r];
  }
}

// ---------------------------------------------------------------- output projection
__global__ void __launch_bounds__(256)
proj_o_kernel(const __bf16* __restrict__ cx, const __bf16* __restrict__ wo,
              float* __restrict__ out) {
  int lane = threadIdx.x & 31, wave = threadIdx.x >> 5;
  int m0 = blockIdx.x * 16;
  int n0 = blockIdx.y * 512 + wave * 64;

  v8f acc[4] = {};
  gemm_16x64(cx + (size_t)m0 * D_, wo, n0, lane, acc);

  int colb = lane & 15, rowgrp = lane >> 4;
#pragma unroll
  for (int j = 0; j < 4; ++j)
#pragma unroll
    for (int r = 0; r < 8; ++r)
      __builtin_nontemporal_store(
          acc[j][r], &out[(size_t)(m0 + r + 8 * rowgrp) * D_ + n0 + j * 16 + colb]);
}

// ---------------------------------------------------------------- launch
extern "C" void kernel_launch(void* const* d_in, const int* in_sizes, int n_in,
                              void* d_out, int out_size, void* d_ws, size_t ws_size,
                              hipStream_t stream) {
  const float* x  = (const float*)d_in[0];
  const float* Wq = (const float*)d_in[1];
  const float* Wk = (const float*)d_in[2];
  const float* Wv = (const float*)d_in[3];
  const float* Wo = (const float*)d_in[4];

  float* out  = (float*)d_out;                        // [B,T,D]
  float* attn = out + (size_t)B_ * T_ * D_;           // [B,H,T,T]

  char* p = (char*)d_ws;
  const size_t N_X = (size_t)B_ * T_ * D_;            // 8,388,608
  const size_t N_W = (size_t)D_ * D_;                 // 1,048,576
  __bf16* x_bf  = (__bf16*)p; p += N_X * 2;
  __bf16* wq_bf = (__bf16*)p; p += N_W * 2;
  __bf16* wk_bf = (__bf16*)p; p += N_W * 2;
  __bf16* wv_bf = (__bf16*)p; p += N_W * 2;
  __bf16* wo_bf = (__bf16*)p; p += N_W * 2;
  __bf16* q_bf  = (__bf16*)p; p += N_X * 2;           // [B,H,T,HD]
  __bf16* k_bf  = (__bf16*)p; p += N_X * 2;           // [B,H,T,HD]
  __bf16* vt_bf = (__bf16*)p; p += N_X * 2;           // [B,H,HD,T]
  __bf16* ctx_bf= (__bf16*)p; p += N_X * 2;           // [B*T, D]

  cvt_f32_to_bf16<<<(int)(N_X / 4 / 256), 256, 0, stream>>>(x,  x_bf,  (int)(N_X / 4));
  cvt_f32_to_bf16<<<(int)(N_W / 4 / 256), 256, 0, stream>>>(Wq, wq_bf, (int)(N_W / 4));
  cvt_f32_to_bf16<<<(int)(N_W / 4 / 256), 256, 0, stream>>>(Wk, wk_bf, (int)(N_W / 4));
  cvt_f32_to_bf16<<<(int)(N_W / 4 / 256), 256, 0, stream>>>(Wv, wv_bf, (int)(N_W / 4));
  cvt_f32_to_bf16<<<(int)(N_W / 4 / 256), 256, 0, stream>>>(Wo, wo_bf, (int)(N_W / 4));

  dim3 gq((B_ * T_) / 16, D_ / 512, 3);               // (512, 2, 3)
  proj_qkv_kernel<<<gq, 256, 0, stream>>>(x_bf, wq_bf, wk_bf, wv_bf,
                                          q_bf, k_bf, vt_bf);

  dim3 ga(T_ / 16, H_, B_);                           // (128, 16, 4)
  attn_kernel<<<ga, 128, 0, stream>>>(q_bf, k_bf, vt_bf, attn, ctx_bf);

  dim3 go((B_ * T_) / 16, D_ / 512, 1);               // (512, 2)
  proj_o_kernel<<<go, 256, 0, stream>>>(ctx_bf, wo_bf, out);
}
